// BaseRNN_78589311582716
// MI455X (gfx1250) — matrix operand order repaired and checked by reading, not compiled, not run on hardware
//
#include <hip/hip_runtime.h>
#include <stdint.h>

// Problem dims (fixed by reference)
#define S_LEN 512
#define BATCH 64
#define EMB_D 512
#define HID_D 512
#define RNN_WGS 16   // persistent workgroups for the recurrent kernel

#define AS_GLOBAL __attribute__((address_space(1)))
#define AS_LOCAL  __attribute__((address_space(3)))

#if __has_builtin(__builtin_amdgcn_global_load_async_to_lds_b128)
#define HAVE_ASYNC_LDS 1
#else
#define HAVE_ASYNC_LDS 0
#endif

#if __has_builtin(__builtin_amdgcn_tensor_load_to_lds) && __has_builtin(__builtin_amdgcn_s_wait_tensorcnt)
#define HAVE_TDM 1
#else
#define HAVE_TDM 0
#endif

#if __has_builtin(__builtin_amdgcn_s_cluster_barrier)
#define HAVE_CLUSTER_BAR 1
#else
#define HAVE_CLUSTER_BAR 0
#endif

typedef __bf16 bf16_t;
typedef __attribute__((ext_vector_type(16))) bf16_t v16bf;
typedef __attribute__((ext_vector_type(8)))  float  v8f;

#define WMMA_BF16(a, b, c) \
  __builtin_amdgcn_wmma_f32_16x16x32_bf16(false, (a), false, (b), (short)0, (c), false, false)

__device__ __forceinline__ unsigned short f2bf(float f) {
  union { float f; unsigned int u; } c; c.f = f;
  unsigned int r = 0x7FFFu + ((c.u >> 16) & 1u);   // round-to-nearest-even
  return (unsigned short)((c.u + r) >> 16);
}

__device__ __forceinline__ void wait_async0() {
#if __has_builtin(__builtin_amdgcn_s_wait_asynccnt)
  __builtin_amdgcn_s_wait_asynccnt(0);
#else
  asm volatile("s_wait_asynccnt 0" ::: "memory");
#endif
}

#if HAVE_ASYNC_LDS
// Revealed signature: param0 = global int4* (src), param1 = local int4* (dst),
// then imm offset, imm cpol.
typedef int v4i_vs __attribute__((vector_size(16)));
__device__ __forceinline__ void async_g2l_b128(const AS_GLOBAL char* src, AS_LOCAL char* dst) {
  __builtin_amdgcn_global_load_async_to_lds_b128((AS_GLOBAL v4i_vs*)src,
                                                 (AS_LOCAL v4i_vs*)dst, 0, 0);
}
#endif

union FragU { v16bf v; uint4 q[2]; };

// A fragment: 16x32 bf16 (M x K), row-major source. ISA wave32 layout:
// lanes 0-15: M=lane, K {0..7, 16..23}; lanes 16-31: K {8..15, 24..31}.
__device__ __forceinline__ v16bf load_a_frag(const unsigned short* __restrict__ A,
                                             int row0, int k0, int ld) {
  int lane = threadIdx.x & 31;
  int m  = lane & 15;
  int kh = lane >> 4;
  const unsigned short* p = A + (size_t)(row0 + m) * ld + k0 + kh * 8;
  FragU u;
  u.q[0] = *reinterpret_cast<const uint4*>(p);
  u.q[1] = *reinterpret_cast<const uint4*>(p + 16);
  return u.v;
}

// B fragment: 32x16 (K x N) with B[k][n] = W[n0+n][k0+k], W row-major (n,k).
__device__ __forceinline__ v16bf load_b_frag(const unsigned short* __restrict__ W,
                                             int n0, int k0, int ld) {
  int lane = threadIdx.x & 31;
  int n  = lane & 15;
  int kh = lane >> 4;
  const unsigned short* p = W + (size_t)(n0 + n) * ld + k0 + kh * 8;
  FragU u;
  u.q[0] = *reinterpret_cast<const uint4*>(p);
  u.q[1] = *reinterpret_cast<const uint4*>(p + 16);
  return u.v;
}

#if HAVE_TDM
typedef __attribute__((ext_vector_type(4))) unsigned int v4u;
typedef __attribute__((ext_vector_type(8))) int          v8i;
typedef __attribute__((ext_vector_type(4))) int          v4i;

// 1-D TDM load: nElem8 elements of 8 bytes, contiguous, global -> LDS.
// D# per ISA ch.8: group0 = {flags(count=1), lds_addr, global_addr, type=2};
// group1 packs data_size=8B, tensor_dim0 = tile_dim0 = nElem8, stride = nElem8.
__device__ __forceinline__ void tdm_load_1d(unsigned lds_byte_addr, const void* gsrc,
                                            unsigned nElem8) {
  unsigned long long ga = (unsigned long long)gsrc;
  v4u g0 = { 1u,                                   // count=1, user descriptor
             lds_byte_addr,
             (unsigned)ga,
             (unsigned)((ga >> 32) & 0x1FFFFFFu) | (2u << 30) };  // type=2 (image)
  v8i g1;
  g1[0] = 0x00030000;                              // workgroup_mask=0, data_size=3 (8B)
  g1[1] = (int)((nElem8 & 0xFFFFu) << 16);         // tensor_dim0[15:0]
  g1[2] = (int)((nElem8 >> 16) | (1u << 16));      // tensor_dim0[31:16] | tensor_dim1=1
  g1[3] = (int)((nElem8 & 0xFFFFu) << 16);         // tile_dim0
  g1[4] = 0;                                       // tile_dim1 = tile_dim2 = 0 (unused)
  g1[5] = (int)nElem8;                             // tensor_dim0_stride[31:0]
  g1[6] = 0;
  g1[7] = 0;
  v4i z4 = {0, 0, 0, 0};
#if __clang_major__ >= 23
  v8i z8 = {0, 0, 0, 0, 0, 0, 0, 0};
  __builtin_amdgcn_tensor_load_to_lds(g0, g1, z4, z4, z8, 0);
#else
  __builtin_amdgcn_tensor_load_to_lds(g0, g1, z4, z4, 0);
#endif
}
#endif  // HAVE_TDM

// ---------------------------------------------------------------------------
// Kernel 1: fp32 -> bf16 weight conversion + bias folding (b_ih + b_hh)
// ---------------------------------------------------------------------------
__global__ void convert_weights_kernel(const float* __restrict__ W_ih, const float* __restrict__ W_hh,
                                       const float* __restrict__ b_ih, const float* __restrict__ b_hh,
                                       unsigned short* __restrict__ Wih_bf,
                                       unsigned short* __restrict__ Whh_bf,
                                       float* __restrict__ bias2) {
  size_t i = (size_t)blockIdx.x * blockDim.x + threadIdx.x;
  const size_t total = (size_t)2 * HID_D * EMB_D;
  if (i < total) {
    Wih_bf[i] = f2bf(W_ih[i]);
    Whh_bf[i] = f2bf(W_hh[i]);
  }
  if (i < 2 * HID_D) bias2[i] = b_ih[i] + b_hh[i];
}

// ---------------------------------------------------------------------------
// Kernel 2: embedding gather + bf16 convert; xe laid out (s, b, e) row-major
// ---------------------------------------------------------------------------
__global__ void embed_gather_kernel(const int* __restrict__ x, const float* __restrict__ emb,
                                    unsigned short* __restrict__ xe) {
  int r = blockIdx.x;                 // r = s*B + b
  int s = r / BATCH;
  int b = r - s * BATCH;
  int tok = x[(size_t)b * S_LEN + s]; // x is (B, S)
  const float* src = emb + (size_t)tok * EMB_D;
  unsigned short* dst = xe + (size_t)r * EMB_D;
  for (int e = threadIdx.x; e < EMB_D; e += blockDim.x) dst[e] = f2bf(src[e]);
}

// ---------------------------------------------------------------------------
// Kernel 3: big hoisted input GEMM  C[M x 512] = A_bf @ W_bfᵀ + bias
// WG tile 64x128, 8 waves, each wave a 32x32 block = 2x2 WMMA tiles.
// ---------------------------------------------------------------------------
__global__ __launch_bounds__(256) void gemm_bias_kernel(
    const unsigned short* __restrict__ A,   // M x 512 bf16
    const unsigned short* __restrict__ W,   // 512 x 512 bf16, (n, k) row-major
    const float* __restrict__ bias,         // 512
    float* __restrict__ C, int M) {
  int wave = (int)(threadIdx.x >> 5);
  int wm = wave & 1;
  int wn = wave >> 1;
  int row0 = blockIdx.x * 64  + wm * 32;
  int col0 = blockIdx.y * 128 + wn * 32;
  if (row0 >= M) return;

  v8f acc00 = {}, acc01 = {}, acc10 = {}, acc11 = {};
  for (int k0 = 0; k0 < EMB_D; k0 += 32) {
    v16bf a0 = load_a_frag(A, row0,      k0, EMB_D);
    v16bf a1 = load_a_frag(A, row0 + 16, k0, EMB_D);
    v16bf b0 = load_b_frag(W, col0,      k0, EMB_D);
    v16bf b1 = load_b_frag(W, col0 + 16, k0, EMB_D);
    acc00 = WMMA_BF16(a0, b0, acc00);
    acc01 = WMMA_BF16(a0, b1, acc01);
    acc10 = WMMA_BF16(a1, b0, acc10);
    acc11 = WMMA_BF16(a1, b1, acc11);
  }

  int lane = threadIdx.x & 31;
  int n  = lane & 15;
  int kh = lane >> 4;
  float bs0 = bias[col0 + n];
  float bs1 = bias[col0 + 16 + n];
  v8f accs[2][2] = {{acc00, acc01}, {acc10, acc11}};
  for (int i = 0; i < 2; ++i)
    for (int j = 0; j < 2; ++j) {
      float bj = (j == 0) ? bs0 : bs1;
      #pragma unroll
      for (int v = 0; v < 8; ++v) {
        int row = row0 + i * 16 + kh * 8 + v;   // D layout: M = v + 8*(lane>=16)
        int col = col0 + j * 16 + n;
        C[(size_t)row * HID_D + col] = accs[i][j][v] + bj;
      }
    }
}

// ---------------------------------------------------------------------------
// Kernel 4: persistent recurrent layer.
// 16 WGs; WG wg owns output columns [wg*32, wg*32+32).
// LDS (dynamic, 96KB): [0,32K) W_hh slice, [32K,96K) h_prev staging buffer.
// Per step: TDM (or async/plain) stage h_prev -> LDS; h_new = tanh(X[t] + h@Whhᵀ),
// masked by t < lengths[b]; device-wide barrier (cluster-barrier hint + atomic).
// ---------------------------------------------------------------------------
__global__ __launch_bounds__(256) void rnn_layer_kernel(
    const float* __restrict__ X,              // (S, B, H) fp32, input GEMM + biases
    const unsigned short* __restrict__ Whh,   // (H, H) bf16, (n, k) row-major
    const int* __restrict__ lengths,          // (B)
    unsigned short* __restrict__ hbuf,        // 2 * B*H bf16 double buffer (zeroed)
    float* __restrict__ h32,                  // B*H fp32 running hidden (zeroed)
    unsigned short* __restrict__ hall,        // (S, B, H) bf16 layer output, or null
    unsigned int* __restrict__ ctr,           // zeroed monotonic barrier counter
    float* __restrict__ finA, int ldA,        // final-h destination(s)
    float* __restrict__ finB, int ldB) {
  extern __shared__ char smem[];
  unsigned short* Wlds = (unsigned short*)smem;                 // 32 KB
  unsigned short* Hlds = (unsigned short*)(smem + 32 * 1024);   // 64 KB

  int wg = blockIdx.x;
  int n_base = wg * 32;

  // --- one-time stage: 32 rows of W_hh into LDS (ASYNC path if available) ---
#if HAVE_ASYNC_LDS
  {
    const AS_GLOBAL char* src = (const AS_GLOBAL char*)(Whh + (size_t)n_base * HID_D);
    AS_LOCAL char*        dst = (AS_LOCAL char*)Wlds;
    const int n128 = (32 * HID_D * 2) / 16;   // 2048 x b128
    for (int i = threadIdx.x; i < n128; i += blockDim.x)
      async_g2l_b128(src + (size_t)i * 16, dst + (size_t)i * 16);
    wait_async0();
  }
#else
  {
    const uint4* src = reinterpret_cast<const uint4*>(Whh + (size_t)n_base * HID_D);
    uint4* dst = reinterpret_cast<uint4*>(Wlds);
    const int n128 = (32 * HID_D * 2) / 16;
    for (int i = threadIdx.x; i < n128; i += blockDim.x) dst[i] = src[i];
  }
#endif
  __syncthreads();

  int wave = (int)(threadIdx.x >> 5);
  int m  = wave & 3;       // row tile: batch rows 16*m .. 16*m+15
  int nt = wave >> 2;      // 0..1 -> local col tile
  int lane = threadIdx.x & 31;
  int n  = lane & 15;
  int kh = lane >> 4;
  int colg = n_base + nt * 16 + n;

#if HAVE_TDM
  unsigned hlds_addr = (unsigned)(unsigned long long)(AS_LOCAL unsigned short*)Hlds;
#endif

  int rowlen[8];
  #pragma unroll
  for (int v = 0; v < 8; ++v) rowlen[v] = lengths[m * 16 + kh * 8 + v];

  for (int t = 0; t < S_LEN; ++t) {
    const unsigned short* hin = hbuf + (size_t)(t & 1) * (BATCH * HID_D);
    unsigned short* hout      = hbuf + (size_t)((t + 1) & 1) * (BATCH * HID_D);

    // --- stage h_prev (64 KB) into LDS ---
#if HAVE_TDM
    if (wave == 0) {
      tdm_load_1d(hlds_addr, hin, (BATCH * HID_D * 2) / 8);   // one DMA op
      __builtin_amdgcn_s_wait_tensorcnt(0);
    }
#elif HAVE_ASYNC_LDS
    {
      const AS_GLOBAL char* src = (const AS_GLOBAL char*)hin;
      AS_LOCAL char*        dst = (AS_LOCAL char*)Hlds;
      const int n128 = (BATCH * HID_D * 2) / 16;   // 4096 x b128
      for (int i = threadIdx.x; i < n128; i += blockDim.x)
        async_g2l_b128(src + (size_t)i * 16, dst + (size_t)i * 16);
      wait_async0();
    }
#else
    {
      const uint4* src = reinterpret_cast<const uint4*>(hin);
      uint4* dst = reinterpret_cast<uint4*>(Hlds);
      const int n128 = (BATCH * HID_D * 2) / 16;
      for (int i = threadIdx.x; i < n128; i += blockDim.x) dst[i] = src[i];
    }
#endif
    __syncthreads();

    const float* Xt = X + (size_t)t * (BATCH * HID_D);

    // prefetch next step's X slice while the WMMAs run
    if (t + 1 < S_LEN) {
      size_t gnext = (size_t)(m * 16 + kh * 8) * HID_D + colg;
      __builtin_prefetch(Xt + BATCH * HID_D + gnext, 0, 0);
    }

    v8f acc = {};
    #pragma unroll 4
    for (int k0 = 0; k0 < HID_D; k0 += 32) {
      v16bf a = load_a_frag(Hlds, m * 16, k0, HID_D);          // ds_load_b128
      FragU u;                                                 // B frag from LDS
      const unsigned short* p = &Wlds[(size_t)(nt * 16 + n) * HID_D + k0 + kh * 8];
      u.q[0] = *reinterpret_cast<const uint4*>(p);
      u.q[1] = *reinterpret_cast<const uint4*>(p + 16);
      acc = WMMA_BF16(a, u.v, acc);
    }

    unsigned short* hallt = hall ? hall + (size_t)t * (BATCH * HID_D) : nullptr;
    #pragma unroll
    for (int v = 0; v < 8; ++v) {
      int row = m * 16 + kh * 8 + v;              // batch index
      size_t gi = (size_t)row * HID_D + colg;
      float val = tanhf(acc[v] + Xt[gi]);
      float hv  = (t < rowlen[v]) ? val : h32[gi];
      h32[gi] = hv;
      unsigned short hb = f2bf(hv);
      hout[gi] = hb;
      if (hallt) hallt[gi] = hb;
    }

    // device-wide step barrier (16 WGs), monotonic counter -> no reset races
    __threadfence();
    __syncthreads();
#if HAVE_CLUSTER_BAR
    __builtin_amdgcn_s_cluster_barrier();   // fast path if cluster-dispatched; NOP otherwise
#endif
    if (threadIdx.x == 0) {
      atomicAdd(ctr, 1u);
      unsigned int target = (unsigned int)(t + 1) * RNN_WGS;
      while (__hip_atomic_load(ctr, __ATOMIC_ACQUIRE, __HIP_MEMORY_SCOPE_AGENT) < target)
        __builtin_amdgcn_s_sleep(2);
    }
    __syncthreads();
    __threadfence();
  }

  // write final hidden (fp32) to output destinations
  #pragma unroll
  for (int v = 0; v < 8; ++v) {
    int row = m * 16 + kh * 8 + v;
    size_t gi = (size_t)row * HID_D + colg;
    float hv = h32[gi];
    finA[(size_t)row * ldA + colg] = hv;
    if (finB) finB[(size_t)row * ldB + colg] = hv;
  }
}

// ---------------------------------------------------------------------------
// Launcher
// ---------------------------------------------------------------------------
extern "C" void kernel_launch(void* const* d_in, const int* in_sizes, int n_in,
                              void* d_out, int out_size, void* d_ws, size_t ws_size,
                              hipStream_t stream) {
  (void)in_sizes; (void)n_in; (void)out_size; (void)ws_size;

  const int*   x    = (const int*)d_in[0];
  const int*   lens = (const int*)d_in[1];
  const float* emb  = (const float*)d_in[2];
  const float* W_ih = (const float*)d_in[3];
  const float* W_hh = (const float*)d_in[4];
  const float* b_ih = (const float*)d_in[5];
  const float* b_hh = (const float*)d_in[6];
  float* out = (float*)d_out;   // [out(B,H)] ++ [hidden(B,L,H)]

  char* ws = (char*)d_ws;
  size_t off = 0;
  auto take = [&](size_t bytes) -> char* {
    char* p = ws + off;
    off += (bytes + 255) & ~(size_t)255;
    return p;
  };

  unsigned short* Wih_bf = (unsigned short*)take((size_t)2 * HID_D * EMB_D * 2);
  unsigned short* Whh_bf = (unsigned short*)take((size_t)2 * HID_D * HID_D * 2);
  float*          bias2  = (float*)take((size_t)2 * HID_D * 4);
  unsigned short* xe     = (unsigned short*)take((size_t)S_LEN * BATCH * EMB_D * 2); // reused as h0_all
  float*          X      = (float*)take((size_t)S_LEN * BATCH * HID_D * 4);          // reused across layers
  unsigned short* hbuf   = (unsigned short*)take((size_t)2 * BATCH * HID_D * 2);
  float*          h32    = (float*)take((size_t)BATCH * HID_D * 4);
  unsigned int*   ctr    = (unsigned int*)take(256);

  const size_t hbuf_bytes = (size_t)2 * BATCH * HID_D * 2;
  const size_t h32_bytes  = (size_t)BATCH * HID_D * 4;
  const int    rnn_lds    = 96 * 1024;   // 32KB W slice + 64KB h_prev

  // --- init (captured in graph; re-runs every replay -> deterministic) ---
  (void)hipMemsetAsync(ctr, 0, 256, stream);
  (void)hipMemsetAsync(hbuf, 0, hbuf_bytes, stream);
  (void)hipMemsetAsync(h32, 0, h32_bytes, stream);

  // --- precompute: bf16 weights + folded biases; embedding gather ---
  convert_weights_kernel<<<(2 * HID_D * EMB_D + 255) / 256, 256, 0, stream>>>(
      W_ih, W_hh, b_ih, b_hh, Wih_bf, Whh_bf, bias2);
  embed_gather_kernel<<<S_LEN * BATCH, 256, 0, stream>>>(x, emb, xe);

  dim3 gg(S_LEN * BATCH / 64, HID_D / 128);

  // --- layer 0: hoisted input GEMM, then recurrence (writes h0_all into xe) ---
  gemm_bias_kernel<<<gg, 256, 0, stream>>>(xe, Wih_bf, bias2, X, S_LEN * BATCH);
  rnn_layer_kernel<<<RNN_WGS, 256, rnn_lds, stream>>>(
      X, Whh_bf, lens, hbuf, h32, xe, ctr,
      out + BATCH * HID_D /* hidden[:,0] */, 2 * HID_D, nullptr, 0);

  // --- layer 1: reset state, hoisted input GEMM over h0_all, recurrence ---
  (void)hipMemsetAsync(hbuf, 0, hbuf_bytes, stream);
  (void)hipMemsetAsync(h32, 0, h32_bytes, stream);
  gemm_bias_kernel<<<gg, 256, 0, stream>>>(xe, Wih_bf + (size_t)HID_D * EMB_D,
                                           bias2 + HID_D, X, S_LEN * BATCH);
  rnn_layer_kernel<<<RNN_WGS, 256, rnn_lds, stream>>>(
      X, Whh_bf + (size_t)HID_D * HID_D, lens, hbuf, h32, nullptr, ctr + 16,
      out + BATCH * HID_D + HID_D /* hidden[:,1] */, 2 * HID_D,
      out /* out duplicates last layer */, HID_D);
}
